// GraphAttentionLayer_71416716198360
// MI455X (gfx1250) — matrix operand order repaired and checked by reading, not compiled, not run on hardware
//
#include <hip/hip_runtime.h>

typedef __attribute__((ext_vector_type(16))) __bf16 v16bf;
typedef __attribute__((ext_vector_type(8)))  __bf16 v8bf;
typedef __attribute__((ext_vector_type(8)))  float  v8f;
typedef __attribute__((ext_vector_type(4)))  unsigned int v4u;
typedef __attribute__((ext_vector_type(8)))  int v8i;
typedef __attribute__((ext_vector_type(4)))  int v4i;

#define BN 16
#define FN 128
#define NN 256
#define TN 64
#define CN 64
#define NT (NN*TN)            // 16384
#define GAT_NEG_INF (-9000000000000000.0f)
#define GAT_ALPHA   0.2f

static __device__ __forceinline__ v16bf make_v16(v8bf lo, v8bf hi) {
  union { v16bf v; v8bf h[2]; } u; u.h[0] = lo; u.h[1] = hi; return u.v;
}

// ---------------------------------------------------------------------------
// Kernel A: Wh = W^T x h per batch (M=64, K=128, N=16384), bf16 WMMA, fp32 acc.
//   - f/g computed in-register (cross-half __shfl_xor reduction)
//   - epilogue transposes tiles through LDS and stores Wh as bf16 in
//     (B,T,C,N) layout -> kernel B's per-(b,t) slice is one contiguous 32 KB
//     block (kills the 17 GB L2 fetch-amplification of the strided layout)
// Grid: (NT/256, B), 256 threads. Each wave: 2 nt-tiles x 4 c-tiles.
// ---------------------------------------------------------------------------
__global__ __launch_bounds__(256) void gat_wh_kernel(
    const float* __restrict__ h, const float* __restrict__ W,
    const float* __restrict__ a,
    __bf16* __restrict__ Whb, float* __restrict__ fbuf, float* __restrict__ gbuf)
{
  __shared__ __align__(16) unsigned char smem[38400];
  __bf16* Wlds = (__bf16*)smem;              // [64][136] W^T, K-fast
  __bf16* hlds = (__bf16*)(smem + 17408);    // [256][40] h chunk, K-fast
  float*  alds = (float*)(smem + 37888);     // [128] a1|a2
  __bf16* tbuf = (__bf16*)smem;              // [64][256] epilogue alias

  const int b    = blockIdx.y;
  const int nt0  = blockIdx.x * 256;
  const int m0   = nt0 >> 6;                 // node base: block = 4 nodes x 64 t
  const int tid  = threadIdx.x;
  const int lane = tid & 31;
  const int wave = tid >> 5;
  const int hi   = lane >> 4;
  const int l15  = lane & 15;

  for (int idx = tid; idx < 128 * 64; idx += 256) {
    const int c = idx & 63, f = idx >> 6;
    Wlds[c * 136 + f] = (__bf16)W[(b * 128 + f) * 64 + c];
  }
  if (tid < 128) alds[tid] = a[b * 128 + tid];

  v8f acc[4][2] = {};

  for (int kc = 0; kc < 4; ++kc) {
    const int kb = kc * 32;
    __syncthreads();
    for (int idx = tid; idx < 32 * 256; idx += 256) {
      const int ntl = idx & 255, fo = idx >> 8;
      hlds[ntl * 40 + fo] =
          (__bf16)h[(size_t)(b * 128 + kb + fo) * NT + nt0 + ntl];
    }
    __syncthreads();

    v16bf afrag[4];
    for (int ct = 0; ct < 4; ++ct) {
      const __bf16* p = &Wlds[(ct * 16 + l15) * 136 + kb + hi * 8];
      afrag[ct] = make_v16(*(const v8bf*)p, *(const v8bf*)(p + 16));
    }
    for (int j = 0; j < 2; ++j) {
      const int ntl = (2 * wave + j) * 16 + l15;
      const __bf16* q = &hlds[ntl * 40 + hi * 8];
      const v16bf bfrag = make_v16(*(const v8bf*)q, *(const v8bf*)(q + 16));
      for (int ct = 0; ct < 4; ++ct)
        acc[ct][j] = __builtin_amdgcn_wmma_f32_16x16x32_bf16(
            false, afrag[ct], false, bfrag, (short)0, acc[ct][j], false, false);
    }
  }

  // f/g reductions; store transposed as (B,T,N) so kernel B reads coalesced
  for (int j = 0; j < 2; ++j) {
    float pf = 0.f, pg = 0.f;
    for (int ct = 0; ct < 4; ++ct)
      for (int r = 0; r < 8; ++r) {
        const int c = ct * 16 + r + 8 * hi;
        pf += acc[ct][j][r] * alds[c];
        pg += acc[ct][j][r] * alds[64 + c];
      }
    pf += __shfl_xor(pf, 16);
    pg += __shfl_xor(pg, 16);
    if (lane < 16) {
      const int loc = (2 * wave + j) * 16 + lane;   // local nt
      const int tt = loc & 63, nl = loc >> 6;
      fbuf[(b * 64 + tt) * 256 + m0 + nl] = pf;
      gbuf[(b * 64 + tt) * 256 + m0 + nl] = pg;
    }
  }

  // Transpose D tiles through LDS, emit packed 8B stores in (B,T,C,N)
  __syncthreads();   // all WMMA fragment reads done; safe to alias smem
  for (int j = 0; j < 2; ++j) {
    const int loc = (2 * wave + j) * 16 + l15;
    for (int ct = 0; ct < 4; ++ct)
      for (int r = 0; r < 8; ++r)
        tbuf[(ct * 16 + r + 8 * hi) * 256 + loc] = (__bf16)acc[ct][j][r];
  }
  __syncthreads();
  for (int it = 0; it < 16; ++it) {
    const int idx = it * 256 + tid;
    const int tt = idx >> 6, c = idx & 63;
    union { __bf16 e[4]; uint2 v; } pk;
    for (int k = 0; k < 4; ++k) pk.e[k] = tbuf[c * 256 + k * 64 + tt];
    *(uint2*)&Whb[((size_t)(b * 64 + tt) * 64 + c) * 256 + m0] = pk.v;
  }
}

// ---------------------------------------------------------------------------
// Kernel B: per (b, t, 32-row n-block):
//   - wave 0 issues ONE tensor_load_to_lds DMA of the contiguous 32 KB
//     Wh^T slice, with D# padding (16B per 512B) giving the 264-element
//     bank-spread row stride for free; DMA overlaps the softmax phase
//   - att rows: mask -> leakyrelu -> exp(x-rowmax), unnormalized bf16 in LDS
//   - D = att x Wh^T via 8 chained WMMAs (K=256); /rowsum + ELU in epilogue
// Grid: (8, T, B), 256 threads. Each wave: one 16x16 output tile.
// ---------------------------------------------------------------------------
__global__ __launch_bounds__(256) void gat_attn_kernel(
    const int* __restrict__ adj, const __bf16* __restrict__ Whb,
    const float* __restrict__ fbuf, const float* __restrict__ gbuf,
    float* __restrict__ out)
{
  __shared__ __align__(16) __bf16 whlds[64 * 264];   // TDM dest [c][m], padded
  __shared__ __align__(16) __bf16 attlds[32 * 264];  // [n][m]
  __shared__ float glds[256];
  __shared__ float rowsum[32];

  const int nblk = blockIdx.x;   // 0..7
  const int t    = blockIdx.y;   // 0..63
  const int b    = blockIdx.z;   // 0..15
  const int tid  = threadIdx.x;
  const int lane = tid & 31;
  const int wave = tid >> 5;
  const int hi   = lane >> 4;
  const int l15  = lane & 15;

  glds[tid] = gbuf[(b * 64 + t) * 256 + tid];        // coalesced

  if (tid < 32) {  // one TDM issue per workgroup (TDM ignores EXEC)
    const unsigned lds0 = (unsigned)(uintptr_t)(&whlds[0]);
    const unsigned long long ga =
        (unsigned long long)(uintptr_t)(Whb + (size_t)(b * 64 + t) * (64 * 256));
    // D# group0: count=1 | lds_addr | global_addr[56:0] | type=2
    v4u g0 = { 1u, lds0, (unsigned)ga,
               (unsigned)((ga >> 32) & 0x01FFFFFFu) | 0x80000000u };
    // D# group1: data_size=2B | pad_enable, pad 4 dwords every 128 dwords
    // (row stride 512B+16B = 264 bf16) | tensor_dim0=16384, tensor_dim1=1,
    // tile_dim0=16384, tile_dim1=1, tensor_dim0_stride=16384
    v8i g1 = { 0x07910000, 0x40000000, 0x00010000, 0x40000000,
               1, 0x4000, 0, 0 };
    v4i gz4 = { 0, 0, 0, 0 };
    v8i gz8 = { 0, 0, 0, 0, 0, 0, 0, 0 };
    __builtin_amdgcn_tensor_load_to_lds(g0, g1, gz4, gz4, gz8, 0);
  }

  const int row = tid >> 3;      // 0..31
  const int sub = tid & 7;       // 8 threads per row, 32 m each
  const int n   = nblk * 32 + row;
  const float fn = fbuf[(b * 64 + t) * 256 + n];
  const int* adjrow = adj + (b * 256 + n) * 256;
  __syncthreads();

  float lmax = -3.4e38f;
  for (int i = 0; i < 32; ++i) {
    const int m = sub * 32 + i;
    const float x = fn + glds[m];
    const float e = adjrow[m] > 0 ? (x > 0.f ? x : GAT_ALPHA * x) : GAT_NEG_INF;
    lmax = fmaxf(lmax, e);
  }
  lmax = fmaxf(lmax, __shfl_xor(lmax, 1));
  lmax = fmaxf(lmax, __shfl_xor(lmax, 2));
  lmax = fmaxf(lmax, __shfl_xor(lmax, 4));

  float lsum = 0.f;
  for (int i = 0; i < 32; ++i) {
    const int m = sub * 32 + i;
    const float x = fn + glds[m];
    const float e = adjrow[m] > 0 ? (x > 0.f ? x : GAT_ALPHA * x) : GAT_NEG_INF;
    const float p = __expf(e - lmax);   // all-masked row -> p==1, matches ref
    attlds[row * 264 + m] = (__bf16)p;
    lsum += p;
  }
  lsum += __shfl_xor(lsum, 1);
  lsum += __shfl_xor(lsum, 2);
  lsum += __shfl_xor(lsum, 4);
  if (sub == 0) rowsum[row] = lsum;

  if (tid < 32) __builtin_amdgcn_s_wait_tensorcnt(0);  // DMA done (wave 0)
  __syncthreads();                                      // publish to all waves

  const int ntile = wave >> 2;   // 0..1
  const int ctile = wave & 3;    // 0..3
  const int arow  = ntile * 16 + l15;
  const int ccol  = ctile * 16 + l15;
  v8f acc = {};
  for (int kc = 0; kc < 8; ++kc) {
    const int kb = kc * 32;
    const __bf16* pa = &attlds[arow * 264 + kb + hi * 8];
    const v16bf afrag = make_v16(*(const v8bf*)pa, *(const v8bf*)(pa + 16));
    const __bf16* pb = &whlds[ccol * 264 + kb + hi * 8];
    const v16bf bfrag = make_v16(*(const v8bf*)pb, *(const v8bf*)(pb + 16));
    acc = __builtin_amdgcn_wmma_f32_16x16x32_bf16(
        false, afrag, false, bfrag, (short)0, acc, false, false);
  }

  for (int r = 0; r < 8; ++r) {
    const int rr = ntile * 16 + r + 8 * hi;   // 0..31
    const int ng = nblk * 32 + rr;
    float v = acc[r] / rowsum[rr];            // deferred softmax normalization
    v = v > 0.f ? v : (__expf(v) - 1.f);      // ELU
    out[((size_t)(b * 64 + ccol) * 256 + ng) * 64 + t] = v;
  }
}

// ---------------------------------------------------------------------------
extern "C" void kernel_launch(void* const* d_in, const int* in_sizes, int n_in,
                              void* d_out, int out_size, void* d_ws, size_t ws_size,
                              hipStream_t stream) {
  const float* h   = (const float*)d_in[0];
  const int*   adj = (const int*)d_in[1];
  const float* W   = (const float*)d_in[2];
  const float* a   = (const float*)d_in[3];
  float* out = (float*)d_out;

  __bf16* Whb = (__bf16*)d_ws;                                   // (B,T,C,N) bf16
  const size_t whbytes = (size_t)BN * CN * NT * sizeof(__bf16);  // 33.5 MB
  float* fbuf = (float*)((char*)d_ws + whbytes);                 // (B,T,N)
  float* gbuf = fbuf + (size_t)BN * NT;                          // (B,T,N)

  dim3 gA(NT / 256, BN);
  gat_wh_kernel<<<gA, 256, 0, stream>>>(h, W, a, Whb, fbuf, gbuf);

  dim3 gB(NN / 32, TN, BN);
  gat_attn_kernel<<<gB, 256, 0, stream>>>(adj, Whb, fbuf, gbuf, out);
}
